// PoolLayer_26774826123433
// MI455X (gfx1250) — matrix operand order repaired and checked by reading, not compiled, not run on hardware
//
#include <hip/hip_runtime.h>

// ---------- types ----------
typedef __attribute__((ext_vector_type(8)))  float          v8f;
typedef __attribute__((ext_vector_type(16))) __bf16         v16bf;
typedef __attribute__((ext_vector_type(16))) unsigned short v16us;

#define NB   16
#define DD   128
#define HH   384
#define CHNK 16
#define SW_STRIDE 136   // 128 + 8 ushort pad -> 68-dword row stride, conflict-free b128 LDS reads

// ---------- phase 1: per-chunk top2 + argmax over rows (column-wise, coalesced over d) ----------
__global__ __launch_bounds__(128) void reduce_partial_kernel(
    const float* __restrict__ x, int L, int rowsPerChunk, int CH,
    float* __restrict__ pTop1, float* __restrict__ pTop2, int* __restrict__ pAmax) {
  const int n  = blockIdx.x / CH;
  const int ch = blockIdx.x % CH;
  const int d  = threadIdx.x;
  const int r0 = ch * rowsPerChunk;
  const float* base = x + ((size_t)n * L + r0) * DD + d;
  float m1 = -__builtin_inff(), m2 = -__builtin_inff();
  int am = r0;
  for (int r = 0; r < rowsPerChunk; ++r) {
    float v = base[(size_t)r * DD];
    if (v > m1)      { m2 = m1; m1 = v; am = r0 + r; }  // strict > keeps first index (jnp.argmax)
    else if (v > m2) { m2 = v; }
  }
  const size_t o = ((size_t)n * CH + ch) * DD + d;
  pTop1[o] = m1; pTop2[o] = m2; pAmax[o] = am;
}

// ---------- phase 2: combine chunk partials into exact top1/second/argmax ----------
__global__ __launch_bounds__(128) void reduce_combine_kernel(
    const float* __restrict__ pTop1, const float* __restrict__ pTop2,
    const int* __restrict__ pAmax, int CH,
    float* __restrict__ top1, float* __restrict__ second, int* __restrict__ amax) {
  const int n = blockIdx.x;
  const int d = threadIdx.x;
  float m1 = -__builtin_inff(), m2 = -__builtin_inff();
  int am = 0;
  for (int ch = 0; ch < CH; ++ch) {           // ascending -> first-index tie-break preserved
    const size_t o = ((size_t)n * CH + ch) * DD + d;
    const float t1 = pTop1[o], t2 = pTop2[o];
    const int   a  = pAmax[o];
    if (t1 > m1) { m2 = fmaxf(m1, t2); m1 = t1; am = a; }
    else         { m2 = fmaxf(m2, t1); }
  }
  const size_t o = (size_t)n * DD + d;
  top1[o] = m1; second[o] = m2; amax[o] = am;
}

// ---------- phase 3a: split W into bf16 hi/lo planes (native RNE converts) ----------
__global__ __launch_bounds__(256) void convert_w_kernel(
    const float* __restrict__ W0, const float* __restrict__ W1,
    unsigned short* __restrict__ Whi0, unsigned short* __restrict__ Wlo0,
    unsigned short* __restrict__ Whi1, unsigned short* __restrict__ Wlo1) {
  const int i = blockIdx.x * 256 + threadIdx.x;   // 0 .. 2*128*384-1
  const int g = (i >= DD * HH) ? 1 : 0;
  const int j = i - g * DD * HH;
  const float f = (g ? W1 : W0)[j];
  const __bf16 h = (__bf16)f;                 // hardware cvt, RNE
  const __bf16 l = (__bf16)(f - (float)h);
  union { __bf16 b; unsigned short u; } ch, cl;
  ch.b = h; cl.b = l;
  if (g) { Whi1[j] = ch.u; Wlo1[j] = cl.u; } else { Whi0[j] = ch.u; Wlo0[j] = cl.u; }
}

// ---------- phase 3b: per-n constant row: b + top1_x0*W[:,128:256] + top1_x1*W[:,256:384] ----------
__global__ __launch_bounds__(128) void cvec_kernel(
    const float* __restrict__ W0, const float* __restrict__ b0,
    const float* __restrict__ W1, const float* __restrict__ b1,
    const float* __restrict__ top1_0, const float* __restrict__ top1_1,
    float* __restrict__ cvec0, float* __restrict__ cvec1) {
  const int g = blockIdx.x >> 4;
  const int n = blockIdx.x & 15;
  const int o = threadIdx.x;
  const float* W = g ? W1 : W0;
  const float* b = g ? b1 : b0;
  const float* wrow = W + (size_t)o * HH;
  float s = b[o];
  for (int d = 0; d < DD; ++d) {
    s += top1_0[n * DD + d] * wrow[128 + d];
    s += top1_1[n * DD + d] * wrow[256 + d];
  }
  (g ? cvec1 : cvec0)[n * DD + o] = s;
}

// ---------- phase 4: main GEMM  y[row,:] = x[row,:] @ Wa^T + cvec[n,:]  (bf16 hi/lo x3 WMMA) ----------
__global__ __launch_bounds__(256) void gemm_bf16x3_kernel(
    const float* __restrict__ x, const unsigned short* __restrict__ Whi,
    const unsigned short* __restrict__ Wlo, const float* __restrict__ cvec,
    float* __restrict__ y, int L) {
  extern __shared__ unsigned short smem[];
  unsigned short* sWhi = smem;
  unsigned short* sWlo = smem + DD * SW_STRIDE;

  const int tid = threadIdx.x;
  // stage Wa (cols 0..127) hi/lo planes into padded LDS, vectorized 16B copies
  for (int c = tid; c < 2048; c += 256) {           // 2048 chunks of 8 ushort per plane
    const int o  = c >> 4;
    const int db = (c & 15) << 3;
    *(uint4*)(&sWhi[o * SW_STRIDE + db]) = *(const uint4*)(Whi + (size_t)o * HH + db);
    *(uint4*)(&sWlo[o * SW_STRIDE + db]) = *(const uint4*)(Wlo + (size_t)o * HH + db);
  }
  __syncthreads();

  const int wave = tid >> 5;
  const int lane = tid & 31;
  const int half = lane >> 4;
  const int lr   = lane & 15;

  // A rows: lanes 0-15 and 16-31 both map M = lr (documented 16-bit A layout)
  const size_t rowA = (size_t)blockIdx.x * 128 + wave * 16 + lr;
  const float* __restrict__ xrow = x + rowA * DD;

  v8f acc[8];
#pragma unroll
  for (int t = 0; t < 8; ++t) acc[t] = (v8f){0.f,0.f,0.f,0.f,0.f,0.f,0.f,0.f};

  union Frag { v16us us; v16bf bf; uint4 q[2]; };
  Frag Ahi, Alo;

#pragma unroll
  for (int ks = 0; ks < 4; ++ks) {
    const int k0 = ks * 32;
    // A fragment: i=0..15 -> K=k0+16*half+i
    union { float4 q[4]; float f[16]; } av;
    av.q[0] = *(const float4*)(xrow + k0 + 16 * half);
    av.q[1] = *(const float4*)(xrow + k0 + 16 * half + 4);
    av.q[2] = *(const float4*)(xrow + k0 + 16 * half + 8);
    av.q[3] = *(const float4*)(xrow + k0 + 16 * half + 12);
#pragma unroll
    for (int i = 0; i < 16; ++i) {
      const __bf16 h = (__bf16)av.f[i];       // native v_cvt (RNE)
      Ahi.bf[i] = h;
      Alo.bf[i] = (__bf16)(av.f[i] - (float)h);
    }
#pragma unroll
    for (int ct = 0; ct < 8; ++ct) {
      // B fragment: N = ct*16+lr, K = k0 + 16*half + i  (contiguous 16 bf16 in padded LDS row)
      Frag Bhi, Blo;
      const unsigned short* bh = &sWhi[(ct * 16 + lr) * SW_STRIDE + k0 + 16 * half];
      const unsigned short* bl = &sWlo[(ct * 16 + lr) * SW_STRIDE + k0 + 16 * half];
      Bhi.q[0] = *(const uint4*)(bh); Bhi.q[1] = *(const uint4*)(bh + 8);
      Blo.q[0] = *(const uint4*)(bl); Blo.q[1] = *(const uint4*)(bl + 8);
      acc[ct] = __builtin_amdgcn_wmma_f32_16x16x32_bf16(false, Ahi.bf, false, Bhi.bf,
                                                        (short)0, acc[ct], false, false);
      acc[ct] = __builtin_amdgcn_wmma_f32_16x16x32_bf16(false, Ahi.bf, false, Blo.bf,
                                                        (short)0, acc[ct], false, false);
      acc[ct] = __builtin_amdgcn_wmma_f32_16x16x32_bf16(false, Alo.bf, false, Bhi.bf,
                                                        (short)0, acc[ct], false, false);
    }
  }

  // epilogue: C/D layout -> M = v + 8*half, N = lr ; add per-n constant row
  const int    n     = (int)(((size_t)blockIdx.x * 128) / (size_t)L);
  const size_t rbase = (size_t)blockIdx.x * 128 + wave * 16 + half * 8;
#pragma unroll
  for (int ct = 0; ct < 8; ++ct) {
    const int   col = ct * 16 + lr;
    const float cv  = cvec[n * DD + col];
#pragma unroll
    for (int v = 0; v < 8; ++v) {
      y[(rbase + v) * DD + col] = acc[ct][v] + cv;
    }
  }
}

// ---------- phase 5: sparse rank-1 corrections at the argmax rows ----------
__global__ __launch_bounds__(128) void scatter_correction_kernel(
    const float* __restrict__ W, int colBase,
    const float* __restrict__ top1, const float* __restrict__ second,
    const int* __restrict__ amax, float* __restrict__ y, int L) {
  const int n = blockIdx.x >> 7;
  const int d = blockIdx.x & 127;
  const int o = threadIdx.x;
  const size_t nd = (size_t)n * DD + d;
  const float delta = second[nd] - top1[nd];   // 0 on duplicate max -> exact vs reference
  const int   l     = amax[nd];
  atomicAdd(&y[(((size_t)n * L) + l) * DD + o], delta * W[(size_t)o * HH + colBase + d]);
}

extern "C" void kernel_launch(void* const* d_in, const int* in_sizes, int n_in,
                              void* d_out, int out_size, void* d_ws, size_t ws_size,
                              hipStream_t stream) {
  (void)in_sizes; (void)n_in; (void)out_size; (void)ws_size;
  const float* x0 = (const float*)d_in[0];
  const float* x1 = (const float*)d_in[1];
  const float* W0 = (const float*)d_in[2];
  const float* b0 = (const float*)d_in[3];
  const float* W1 = (const float*)d_in[4];
  const float* b1 = (const float*)d_in[5];
  const int L0 = 2048, L1 = 1024;
  float* y0 = (float*)d_out;
  float* y1 = y0 + (size_t)NB * L0 * DD;

  // workspace carve-out (~1.2 MB), 256B aligned slices
  char* w = (char*)d_ws;
  size_t off = 0;
  auto take = [&](size_t bytes) -> char* {
    char* p = w + off;
    off += (bytes + 255) & ~(size_t)255;
    return p;
  };
  const size_t partial = (size_t)NB * CHNK * DD;      // 32768 elems
  float* pT1_0 = (float*)take(partial * 4);
  float* pT2_0 = (float*)take(partial * 4);
  int*   pAm_0 = (int*)  take(partial * 4);
  float* pT1_1 = (float*)take(partial * 4);
  float* pT2_1 = (float*)take(partial * 4);
  int*   pAm_1 = (int*)  take(partial * 4);
  float* t1_0 = (float*)take((size_t)NB * DD * 4);
  float* s_0  = (float*)take((size_t)NB * DD * 4);
  int*   am_0 = (int*)  take((size_t)NB * DD * 4);
  float* t1_1 = (float*)take((size_t)NB * DD * 4);
  float* s_1  = (float*)take((size_t)NB * DD * 4);
  int*   am_1 = (int*)  take((size_t)NB * DD * 4);
  unsigned short* Whi0 = (unsigned short*)take((size_t)DD * HH * 2);
  unsigned short* Wlo0 = (unsigned short*)take((size_t)DD * HH * 2);
  unsigned short* Whi1 = (unsigned short*)take((size_t)DD * HH * 2);
  unsigned short* Wlo1 = (unsigned short*)take((size_t)DD * HH * 2);
  float* cvec0 = (float*)take((size_t)NB * DD * 4);
  float* cvec1 = (float*)take((size_t)NB * DD * 4);

  // 1) split top2/argmax reductions
  reduce_partial_kernel<<<NB * CHNK, 128, 0, stream>>>(x0, L0, L0 / CHNK, CHNK, pT1_0, pT2_0, pAm_0);
  reduce_partial_kernel<<<NB * CHNK, 128, 0, stream>>>(x1, L1, L1 / CHNK, CHNK, pT1_1, pT2_1, pAm_1);
  reduce_combine_kernel<<<NB, 128, 0, stream>>>(pT1_0, pT2_0, pAm_0, CHNK, t1_0, s_0, am_0);
  reduce_combine_kernel<<<NB, 128, 0, stream>>>(pT1_1, pT2_1, pAm_1, CHNK, t1_1, s_1, am_1);
  // 2) weight split + per-n constant rows
  convert_w_kernel<<<(2 * DD * HH) / 256, 256, 0, stream>>>(W0, W1, Whi0, Wlo0, Whi1, Wlo1);
  cvec_kernel<<<32, 128, 0, stream>>>(W0, b0, W1, b1, t1_0, t1_1, cvec0, cvec1);
  // 3) WMMA GEMMs (K=128 only, concat never materialized)
  const size_t shbytes = (size_t)2 * DD * SW_STRIDE * sizeof(unsigned short);  // 69632 B
  gemm_bf16x3_kernel<<<(NB * L0) / 128, 256, shbytes, stream>>>(x0, Whi0, Wlo0, cvec0, y0, L0);
  gemm_bf16x3_kernel<<<(NB * L1) / 128, 256, shbytes, stream>>>(x1, Whi1, Wlo1, cvec1, y1, L1);
  // 4) sparse loo corrections
  scatter_correction_kernel<<<NB * DD, 128, 0, stream>>>(W0, 128, t1_0, s_0, am_0, y0, L0);
  scatter_correction_kernel<<<NB * DD, 128, 0, stream>>>(W1, 256, t1_1, s_1, am_1, y1, L1);
}
